// SessionSpatialProjector_47356309406352
// MI455X (gfx1250) — matrix operand order repaired and checked by reading, not compiled, not run on hardware
//
#include <hip/hip_runtime.h>
#include <hip/hip_bf16.h>

// ---- problem constants (match reference) ----
#define B_SZ   64
#define MAXC   128
#define MAXT   4096
#define NSRC   256

// ---- tiling ----
#define TILE_N 256   // timesteps per block
#define TILE_M 64    // sources per block
#define XS_K   136   // LDS row stride (elements) for x tile  (128 + pad, 272B = 16B aligned)
#define WS_K   136   // LDS row stride (elements) for W tile

typedef __attribute__((ext_vector_type(16))) __bf16 v16bf;
typedef __attribute__((ext_vector_type(8)))  __bf16 v8bf;
typedef __attribute__((ext_vector_type(8)))  float  v8f;

union PackBF4 { __bf16 h[4]; unsigned long long u64; };

__global__ __launch_bounds__(256)
void session_proj_wmma_bf16(const float* __restrict__ x,     // [B, MAXC, MAXT]
                            const float* __restrict__ W,     // [S, NSRC, MAXC]
                            const float* __restrict__ bias,  // [S, NSRC]
                            const int*   __restrict__ sid,   // [B]
                            const int*   __restrict__ ccnt,  // [B]
                            const int*   __restrict__ slen,  // [B]
                            float* __restrict__ out)         // [B, NSRC, MAXT]
{
    extern __shared__ char smem[];
    __bf16* Xs = (__bf16*)smem;                 // [TILE_N][XS_K]  t-major, c contiguous
    __bf16* Ws = Xs + TILE_N * XS_K;            // [TILE_M][WS_K]  m-major, c contiguous
    float*  Bs = (float*)(Ws + TILE_M * WS_K);  // [TILE_M] bias

    const int tid   = threadIdx.x;
    const int bz    = blockIdx.z;                 // batch
    const int mBase = blockIdx.y * TILE_M;        // source-row base
    const int tBase = blockIdx.x * TILE_N;        // time base

    const int sess = sid[bz];
    const int cc   = ccnt[bz];
    const int len  = slen[bz];

    // ---------- fast path: whole time tile beyond seq_len -> zeros ----------
    if (tBase >= len) {
        const float4 z4 = make_float4(0.f, 0.f, 0.f, 0.f);
        // 64 rows x 256 cols = 4096 float4
        for (int i = tid; i < TILE_M * (TILE_N / 4); i += 256) {
            const int r = i >> 6;          // row within tile
            const int q = i & 63;          // float4 within row
            float4* o4 = (float4*)(out + ((size_t)bz * NSRC + mBase + r) * MAXT + tBase);
            o4[q] = z4;
        }
        return;
    }

    const int kRows  = (cc + 31) & ~31;   // channels rounded up to K-step (x/W are zero-padded)
    const int kSteps = kRows >> 5;        // 2..4

    // ---------- stage W tile (fp32 -> bf16), m-major ----------
    {
        const float* wsrc = W + ((size_t)sess * NSRC + mBase) * MAXC;
        const float4* w4 = (const float4*)wsrc;      // TILE_M*MAXC/4 = 2048 float4
        for (int i = tid; i < TILE_M * (MAXC / 4); i += 256) {
            float4 v = w4[i];
            const int m  = i >> 5;              // 32 float4 per 128-ch row
            const int kq = (i & 31) << 2;
            PackBF4 p;
            p.h[0] = (__bf16)v.x; p.h[1] = (__bf16)v.y;
            p.h[2] = (__bf16)v.z; p.h[3] = (__bf16)v.w;
            *(unsigned long long*)(&Ws[m * WS_K + kq]) = p.u64;
        }
        if (tid < TILE_M) Bs[tid] = bias[(size_t)sess * NSRC + mBase + tid];
    }

    // ---------- stage x tile (fp32 -> bf16), transposed to [t][c] ----------
    {
        const float* xsrc = x + (size_t)bz * MAXC * MAXT + tBase;
        for (int i = tid; i < kRows * (TILE_N / 4); i += 256) {
            const int c  = i >> 6;              // channel row
            const int tq = (i & 63) << 2;       // time within tile
            float4 v = ((const float4*)(xsrc + (size_t)c * MAXT))[i & 63];
            Xs[(tq + 0) * XS_K + c] = (__bf16)v.x;
            Xs[(tq + 1) * XS_K + c] = (__bf16)v.y;
            Xs[(tq + 2) * XS_K + c] = (__bf16)v.z;
            Xs[(tq + 3) * XS_K + c] = (__bf16)v.w;
        }
    }
    __syncthreads();

    // ---------- per-wave WMMA compute: 32(M) x 64(N) per wave ----------
    const int lane  = tid & 31;
    const int l     = lane & 15;       // position within half
    const int half  = lane >> 4;       // 0 / 1
    const int wave  = tid >> 5;        // 0..7
    const int waveM = (wave & 1) * 32; // M offset within block tile
    const int waveN = (wave >> 1) * 64;// N offset within block tile

    v8f acc[2][4];
#pragma unroll
    for (int mt = 0; mt < 2; ++mt)
#pragma unroll
        for (int nt = 0; nt < 4; ++nt)
            acc[mt][nt] = (v8f)(0.0f);

    for (int ks = 0; ks < kSteps; ++ks) {
        const int kb = ks << 5;

        // A tiles: 16x32 bf16. Lane layout: half0 -> K {kb..kb+7, kb+16..kb+23},
        //                                   half1 -> K {kb+8..kb+15, kb+24..kb+31}
        v16bf A[2];
#pragma unroll
        for (int mt = 0; mt < 2; ++mt) {
            const __bf16* ap = &Ws[(waveM + mt * 16 + l) * WS_K + kb + half * 8];
            v8bf a0 = *(const v8bf*)(ap);
            v8bf a1 = *(const v8bf*)(ap + 16);
            A[mt] = __builtin_shufflevector(a0, a1, 0,1,2,3,4,5,6,7,8,9,10,11,12,13,14,15);
        }

        // B tiles: 32x16 bf16. Lane = column; half0 -> K {kb..kb+7, kb+16..kb+23}, half1 -> K {kb+8..kb+15, kb+24..kb+31}
        v16bf Bv[4];
#pragma unroll
        for (int nt = 0; nt < 4; ++nt) {
            const __bf16* bp = &Xs[(waveN + nt * 16 + l) * XS_K + kb + half * 8];
            v8bf b0 = *(const v8bf*)(bp);
            v8bf b1 = *(const v8bf*)(bp + 16);
            Bv[nt] = __builtin_shufflevector(b0, b1, 0,1,2,3,4,5,6,7,8,9,10,11,12,13,14,15);
        }

#pragma unroll
        for (int mt = 0; mt < 2; ++mt)
#pragma unroll
            for (int nt = 0; nt < 4; ++nt)
                acc[mt][nt] = __builtin_amdgcn_wmma_f32_16x16x32_bf16(
                    /*neg_a=*/false, A[mt], /*neg_b=*/false, Bv[nt],
                    /*c_mod=*/(short)0, acc[mt][nt],
                    /*reuse_a=*/false, /*reuse_b=*/false);
    }

    // ---------- bias + time mask + store ----------
    // C/D layout: VGPR j, lane -> M = j + 8*half, N = l
#pragma unroll
    for (int nt = 0; nt < 4; ++nt) {
        const int tcol = tBase + waveN + nt * 16 + l;
        const bool ok  = tcol < len;
#pragma unroll
        for (int mt = 0; mt < 2; ++mt) {
            v8f c = acc[mt][nt];
#pragma unroll
            for (int j = 0; j < 8; ++j) {
                const int mLoc = waveM + mt * 16 + half * 8 + j;
                const float v  = ok ? (c[j] + Bs[mLoc]) : 0.0f;
                out[((size_t)bz * NSRC + mBase + mLoc) * MAXT + tcol] = v;
            }
        }
    }
}

extern "C" void kernel_launch(void* const* d_in, const int* in_sizes, int n_in,
                              void* d_out, int out_size, void* d_ws, size_t ws_size,
                              hipStream_t stream) {
    const float* x    = (const float*)d_in[0];
    const float* W    = (const float*)d_in[1];
    const float* bias = (const float*)d_in[2];
    const int*   sid  = (const int*)d_in[3];
    const int*   ccnt = (const int*)d_in[4];
    const int*   slen = (const int*)d_in[5];
    float*       out  = (float*)d_out;

    dim3 grid(MAXT / TILE_N, NSRC / TILE_M, B_SZ);   // (16, 4, 64)
    dim3 block(256);
    const size_t lds = (size_t)(TILE_N * XS_K + TILE_M * WS_K) * sizeof(__bf16)
                     + TILE_M * sizeof(float);       // 87,296 B
    session_proj_wmma_bf16<<<grid, block, lds, stream>>>(x, W, bias, sid, ccnt, slen, out);
}
